// DecoderRNNTAtt_33054068310288
// MI455X (gfx1250) — compile-verified, matrix operand order checked by reading, not run on hardware
//
#include <hip/hip_runtime.h>

// Problem constants (from reference)
#define B_    8
#define T_    128
#define U_    32
#define EPROJS 1024
#define DUNITS 1024
#define EMB   512
#define ATT_  512
#define JOINT_ 1024
#define ODIM  4096

typedef __attribute__((ext_vector_type(16))) _Float16 v16h;
typedef __attribute__((ext_vector_type(8)))  _Float16 v8h;
typedef __attribute__((ext_vector_type(4)))  _Float16 v4h;
typedef __attribute__((ext_vector_type(8)))  float    v8f;
typedef __attribute__((ext_vector_type(4)))  unsigned int u32x4;
typedef __attribute__((ext_vector_type(8)))  int      i32x8;
typedef __attribute__((ext_vector_type(4)))  int      i32x4;

// Tensor Data Mover availability (probe-verified builtin on both toolchains,
// arity differs: ROCm7.2/clang-22 = 5 args, amdgpu-toolchain/clang-23 = 6)
#if defined(__has_builtin)
#  if __has_builtin(__builtin_amdgcn_tensor_load_to_lds) && \
      __has_builtin(__builtin_amdgcn_s_wait_tensorcnt)
#    define USE_TDM 1
#  endif
#endif
#ifndef USE_TDM
#  define USE_TDM 0
#endif

// ---------------------------------------------------------------------------
// Fragment load helper: CDNA5 16-bit A/B operand layout (ISA 7.12.2):
//   lanes 0-15  : row (lane),    halves 0..7 = K 0..7,   halves 8..15 = K 16..23
//   lanes 16-31 : row (lane-16), halves 0..7 = K 8..15,  halves 8..15 = K 24..31
// LDS row stride 40 halves (80B, padded) kills bank conflicts; sh=(lane>>4)*8.
// ---------------------------------------------------------------------------
__device__ __forceinline__ v16h load_frag(const _Float16* p) {
  v8h lo = *(const v8h*)p;
  v8h hi = *(const v8h*)(p + 16);
  v16h r;
#pragma unroll
  for (int i = 0; i < 8; ++i) { r[i] = lo[i]; r[i + 8] = hi[i]; }
  return r;
}

// Row of NT WMMAs: issue ALL fragment loads first (single ds clause, single
// dscnt drain), then fire the WMMAs back-to-back so the matrix pipe stays fed
// instead of paying an LDS round-trip per WMMA.
template <int NT>
__device__ __forceinline__ void wmma_row(const _Float16* Bbase, int l, int sh,
                                         v16h a, v8f* acc) {
  v16h b[NT];
#pragma unroll
  for (int nt = 0; nt < NT; ++nt)
    b[nt] = load_frag(&Bbase[(nt * 16 + l) * 40 + sh]);
#pragma unroll
  for (int nt = 0; nt < NT; ++nt)
    acc[nt] = __builtin_amdgcn_wmma_f32_16x16x32_f16(false, a, false, b[nt],
                                                     (short)0, acc[nt], false, false);
}

__device__ __forceinline__ v8h cvt_pack8(const float4* s) {
  float4 p0 = s[0], p1 = s[1];
  v8h h;
  h[0] = (_Float16)p0.x; h[1] = (_Float16)p0.y; h[2] = (_Float16)p0.z; h[3] = (_Float16)p0.w;
  h[4] = (_Float16)p1.x; h[5] = (_Float16)p1.y; h[6] = (_Float16)p1.z; h[7] = (_Float16)p1.w;
  return h;
}

__device__ __forceinline__ v8h tanh_pack8(const float4* e, const float4* d) {
  float4 e0 = e[0], d0 = d[0], e1 = e[1], d1 = d[1];
  v8h h;
  h[0] = (_Float16)tanhf(e0.x + d0.x); h[1] = (_Float16)tanhf(e0.y + d0.y);
  h[2] = (_Float16)tanhf(e0.z + d0.z); h[3] = (_Float16)tanhf(e0.w + d0.w);
  h[4] = (_Float16)tanhf(e1.x + d1.x); h[5] = (_Float16)tanhf(e1.y + d1.y);
  h[6] = (_Float16)tanhf(e1.z + d1.z); h[7] = (_Float16)tanhf(e1.w + d1.w);
  return h;
}

__device__ __forceinline__ float sigf(float x) { return 1.0f / (1.0f + expf(-x)); }

#if USE_TDM
// ---------------------------------------------------------------------------
// TDM: DMA a 128-row x 32-col f16 tile of w_jo^T [ODIM][JOINT] into LDS with
// hardware padding reproducing the 40-half LDS row stride:
//   row = 32 halves = 16 DWORDs stored  (pad_interval code 3 -> 16 DWORDs)
//   pad = 8 halves  = 4  DWORDs skipped (pad_amount  code 3 -> 4 DWORDs)
// D# per ISA ch.8: group0 = {count=1, lds_addr, global_addr[56:0], type=2};
// group1 = {data_size=2B, pad_enable, dims, tile dims, stride}.
// ---------------------------------------------------------------------------
__device__ __forceinline__ void tdm_load_b_tile(const _Float16* gsrc, void* ldsDst) {
  unsigned long long ga = (unsigned long long)(size_t)gsrc;
  unsigned lds_off = (unsigned)(size_t)ldsDst;  // low 32 bits = LDS byte offset
  u32x4 g0;
  g0[0] = 1u;                                         // count=1 (valid user D#)
  g0[1] = lds_off;                                    // lds_addr
  g0[2] = (unsigned)(ga & 0xFFFFFFFFu);               // global_addr[31:0]
  g0[3] = (unsigned)((ga >> 32) & 0x01FFFFFFu)        // global_addr[56:32]
        | (2u << 30);                                 // type=2 ("image")
  i32x8 g1;
  g1[0] = (int)((1u << 16)      // data_size = 2 bytes
              | (1u << 20)      // pad_enable
              | (3u << 22)      // pad_interval: 16 DWORDs
              | (3u << 25));    // pad_amount : 4 DWORDs
  g1[1] = (int)((JOINT_ & 0xFFFFu) << 16);            // tensor_dim0 lo16
  g1[2] = (int)(((JOINT_ >> 16) & 0xFFFFu)            // tensor_dim0 hi16
              | ((ODIM & 0xFFFFu) << 16));            // tensor_dim1 lo16
  g1[3] = (int)(((ODIM >> 16) & 0xFFFFu)              // tensor_dim1 hi16
              | (32u << 16));                         // tile_dim0 = 32
  g1[4] = (int)128;                                   // tile_dim1 = 128 (dim2=0)
  g1[5] = (int)JOINT_;                                // tensor_dim0_stride lo32
  g1[6] = 0;                                          // stride0 hi / stride1 lo
  g1[7] = 0;
  i32x4 g2 = (i32x4)0;
  i32x4 g3 = (i32x4)0;
#if __clang_major__ >= 23
  i32x8 g4 = (i32x8)0;
  __builtin_amdgcn_tensor_load_to_lds(g0, g1, g2, g3, g4, 0);
#else
  __builtin_amdgcn_tensor_load_to_lds(g0, g1, g2, g3, 0);
#endif
}
#endif  // USE_TDM

// ---------------------------------------------------------------------------
// Weight conversion kernels (tiny traffic)
// ---------------------------------------------------------------------------
// src [K][N] f32 row-major  ->  dst [N][K] f16 (B^T layout for WMMA B loads)
__global__ void cvtT_f16(const float* __restrict__ src, _Float16* __restrict__ dst,
                         int K, int N) {
  int idx = blockIdx.x * 256 + threadIdx.x;
  int k = idx / N, n = idx % N;
  dst[(size_t)n * K + k] = (_Float16)src[idx];
}

// straight f32 -> f16, 4 elems/thread (W_ih/W_hh already [N][K])
__global__ void cvt4_f16(const float* __restrict__ src, _Float16* __restrict__ dst) {
  size_t i = ((size_t)blockIdx.x * 256 + threadIdx.x) * 4;
  float4 p = *(const float4*)(src + i);
  v4h h;
  h[0] = (_Float16)p.x; h[1] = (_Float16)p.y; h[2] = (_Float16)p.z; h[3] = (_Float16)p.w;
  *(v4h*)(dst + i) = h;
}

// ---------------------------------------------------------------------------
// Generic 128x128-tile GEMM:  C[M][N] = A[M][K](f32) * Bt[N][K](f16)^T + bias
// M,N multiples of 128, K multiple of 32. 256 threads = 8 waves,
// wave w -> rows [16w,16w+16), all 128 cols (8 WMMA accumulators).
// ---------------------------------------------------------------------------
__global__ __launch_bounds__(256) void gemm_a32_128x128(
    const float* __restrict__ A, const _Float16* __restrict__ Bt,
    const float* __restrict__ bias, float* __restrict__ C, int K, int N) {
  __shared__ __align__(16) _Float16 ldsA[128 * 40];
  __shared__ __align__(16) _Float16 ldsB[128 * 40];
  const int tid = threadIdx.x, lane = tid & 31, wave = tid >> 5;
  const int mblk = blockIdx.y << 7, nblk = blockIdx.x << 7;
  const int lr = tid >> 1;            // staging row 0..127
  const int lh = (tid & 1) << 4;      // col half: 0 or 16
  const float*    aRow = A  + (size_t)(mblk + lr) * K + lh;
  const _Float16* bRow = Bt + (size_t)(nblk + lr) * K + lh;
  v8f acc[8] = {};
  for (int k0 = 0; k0 < K; k0 += 32) {
    {
      const float4* s = (const float4*)(aRow + k0);
      *(v8h*)&ldsA[lr * 40 + lh]     = cvt_pack8(s);
      *(v8h*)&ldsA[lr * 40 + lh + 8] = cvt_pack8(s + 2);
    }
    {
      const v8h* s = (const v8h*)(bRow + k0);
      *(v8h*)&ldsB[lr * 40 + lh]     = s[0];
      *(v8h*)&ldsB[lr * 40 + lh + 8] = s[1];
    }
    if (k0 + 32 < K) {
      __builtin_prefetch(aRow + k0 + 32, 0, 0);
      __builtin_prefetch(bRow + k0 + 32, 0, 0);
    }
    __syncthreads();
    const int l = lane & 15, sh = (lane >> 4) << 3;
    v16h a = load_frag(&ldsA[(wave * 16 + l) * 40 + sh]);
    wmma_row<8>(ldsB, l, sh, a, acc);
    __syncthreads();
  }
  const int l = lane & 15, rs = (lane >> 4) << 3;
#pragma unroll
  for (int nt = 0; nt < 8; ++nt) {
    const int col = nblk + nt * 16 + l;
    const float bv = bias ? bias[col] : 0.0f;
#pragma unroll
    for (int j = 0; j < 8; ++j) {
      const int row = mblk + wave * 16 + rs + j;
      C[(size_t)row * N + col] = acc[nt][j] + bv;
    }
  }
}

// ---------------------------------------------------------------------------
// Skinny-M GEMM (M<=16, zero-padded to 16): sequential decoder steps.
// grid.x = N/128; wave w owns cols [16w,16w+16). accumulate=1 -> C += result.
// ---------------------------------------------------------------------------
__global__ __launch_bounds__(256) void gemm_a32_16rows(
    const float* __restrict__ A, const _Float16* __restrict__ Bt,
    const float* __restrict__ bias, float* __restrict__ C,
    int M, int K, int N, int accumulate) {
  __shared__ __align__(16) _Float16 ldsA[16 * 40];
  __shared__ __align__(16) _Float16 ldsB[128 * 40];
  const int tid = threadIdx.x, lane = tid & 31, wave = tid >> 5;
  const int nblk = blockIdx.x << 7;
  const int lr = tid >> 1, lh = (tid & 1) << 4;
  const _Float16* bRow = Bt + (size_t)(nblk + lr) * K + lh;
  v8f acc = {};
  for (int k0 = 0; k0 < K; k0 += 32) {
    if (tid < 128) {
      const int rr = tid >> 3, c4 = (tid & 7) << 2;
      float4 p;
      if (rr < M) p = *(const float4*)(A + (size_t)rr * K + k0 + c4);
      else        p = make_float4(0.f, 0.f, 0.f, 0.f);
      v4h h;
      h[0] = (_Float16)p.x; h[1] = (_Float16)p.y;
      h[2] = (_Float16)p.z; h[3] = (_Float16)p.w;
      *(v4h*)&ldsA[rr * 40 + c4] = h;
    }
    {
      const v8h* s = (const v8h*)(bRow + k0);
      *(v8h*)&ldsB[lr * 40 + lh]     = s[0];
      *(v8h*)&ldsB[lr * 40 + lh + 8] = s[1];
    }
    __syncthreads();
    const int l = lane & 15, sh = (lane >> 4) << 3;
    v16h a = load_frag(&ldsA[l * 40 + sh]);
    v16h b = load_frag(&ldsB[(wave * 16 + l) * 40 + sh]);
    acc = __builtin_amdgcn_wmma_f32_16x16x32_f16(false, a, false, b,
                                                 (short)0, acc, false, false);
    __syncthreads();
  }
  const int l = lane & 15, rs = (lane >> 4) << 3;
  const int col = nblk + wave * 16 + l;
  const float bv = bias ? bias[col] : 0.0f;
#pragma unroll
  for (int j = 0; j < 8; ++j) {
    const int row = rs + j;
    if (row < M) {
      size_t idx = (size_t)row * N + col;
      float v = acc[j] + bv;
      if (accumulate) v += C[idx];
      C[idx] = v;
    }
  }
}

// ---------------------------------------------------------------------------
// Joint output GEMM (dominant: 32768 x 1024 x 4096, 274 GFLOP).
// Row r = (b*T+t)*U+u.  A[r][k] = tanh(je[r>>5][k] + jd[(b<<5)|u][k]) built
// on the fly into LDS (f16); B = w_jo^T f16 [ODIM][JOINT], staged by the
// Tensor Data Mover (double-buffered, TENSORcnt-synchronized) when available.
// ---------------------------------------------------------------------------
__global__ __launch_bounds__(256) void joint_out_gemm(
    const float* __restrict__ je, const float* __restrict__ jd,
    const _Float16* __restrict__ wjoT, const float* __restrict__ bjo,
    float* __restrict__ out) {
  __shared__ __align__(16) _Float16 ldsA[128 * 40];
#if USE_TDM
  __shared__ __align__(16) _Float16 ldsB[2][128 * 40];
#else
  __shared__ __align__(16) _Float16 ldsB1[128 * 40];
#endif
  const int tid = threadIdx.x, lane = tid & 31, wave = tid >> 5;
  const int mblk = blockIdx.y << 7, nblk = blockIdx.x << 7;
  const int lr = tid >> 1, lh = (tid & 1) << 4;
  const int gr = mblk + lr;                                  // global row
  const float* jeRow = je + (size_t)(gr >> 5) * JOINT_ + lh; // b*T+t = r/U
  const float* jdRow = jd + (size_t)(((gr >> 12) << 5) | (gr & 31)) * JOINT_ + lh;
  const _Float16* bRow = wjoT + (size_t)(nblk + lr) * JOINT_ + lh;
  (void)bRow;
  v8f acc[8] = {};
#if USE_TDM
  int buf = 0;
  if (tid < 32)  // one wave issues the DMA (EXEC ignored by tensor ops)
    tdm_load_b_tile(wjoT + (size_t)nblk * JOINT_, &ldsB[0][0]);
#endif
  for (int k0 = 0; k0 < JOINT_; k0 += 32) {
    // stage A tile: fused tanh(je+jd) -> f16 (overlaps with in-flight DMA)
    {
      const float4* se = (const float4*)(jeRow + k0);
      const float4* sd = (const float4*)(jdRow + k0);
      *(v8h*)&ldsA[lr * 40 + lh]     = tanh_pack8(se, sd);
      *(v8h*)&ldsA[lr * 40 + lh + 8] = tanh_pack8(se + 2, sd + 2);
    }
#if USE_TDM
    if (tid < 32) __builtin_amdgcn_s_wait_tensorcnt(0);
    __syncthreads();                       // B tile (and A tile) visible to all
    if ((k0 + 32 < JOINT_) && (tid < 32))  // prefetch next B tile via TDM
      tdm_load_b_tile(wjoT + (size_t)nblk * JOINT_ + (k0 + 32), &ldsB[buf ^ 1][0]);
    const _Float16* Bb = &ldsB[buf][0];
#else
    {
      const v8h* s = (const v8h*)(bRow + k0);
      *(v8h*)&ldsB1[lr * 40 + lh]     = s[0];
      *(v8h*)&ldsB1[lr * 40 + lh + 8] = s[1];
    }
    if (k0 + 32 < JOINT_) __builtin_prefetch(bRow + k0 + 32, 0, 0);
    __syncthreads();
    const _Float16* Bb = ldsB1;
#endif
    const int l = lane & 15, sh = (lane >> 4) << 3;
    v16h a = load_frag(&ldsA[(wave * 16 + l) * 40 + sh]);
    wmma_row<8>(Bb, l, sh, a, acc);
    __syncthreads();
#if USE_TDM
    buf ^= 1;
#endif
  }
  const int l = lane & 15, rs = (lane >> 4) << 3;
#pragma unroll
  for (int nt = 0; nt < 8; ++nt) {
    const int col = nblk + nt * 16 + l;
    const float bv = bjo[col];
#pragma unroll
    for (int j = 0; j < 8; ++j) {
      const int row = mblk + wave * 16 + rs + j;
      out[(size_t)row * ODIM + col] = acc[nt][j] + bv;
    }
  }
}

// ---------------------------------------------------------------------------
// Attention: e[b,t] = sum_a v[a]*tanh(pre[b,t,a]+q[b,a]); masked softmax over t;
// att_c[b,:] = sum_t w[t]*hs[b,t,:].  One block per b, 128 threads (t).
// ---------------------------------------------------------------------------
__global__ __launch_bounds__(128) void att_kernel(
    const float* __restrict__ pre, const float* __restrict__ q,
    const float* __restrict__ v_att, const float* __restrict__ hs,
    const int* __restrict__ hlens, float* __restrict__ att_c) {
  const int b = blockIdx.x, t = threadIdx.x;
  __shared__ float sw[128];
  __shared__ float red[128];
  int hl = hlens[b]; if (hl < 1) hl = 1;
  const float4* pp = (const float4*)(pre + (size_t)(b * T_ + t) * ATT_);
  const float4* qq = (const float4*)(q + (size_t)b * ATT_);
  const float4* vv = (const float4*)v_att;
  float s = 0.f;
  for (int a = 0; a < ATT_ / 4; ++a) {
    float4 p = pp[a], qv = qq[a], va = vv[a];
    s += va.x * tanhf(p.x + qv.x) + va.y * tanhf(p.y + qv.y)
       + va.z * tanhf(p.z + qv.z) + va.w * tanhf(p.w + qv.w);
  }
  const bool valid = (t < hl);
  red[t] = valid ? s : -3.0e38f;
  __syncthreads();
  for (int st = 64; st > 0; st >>= 1) {
    if (t < st) red[t] = fmaxf(red[t], red[t + st]);
    __syncthreads();
  }
  const float m = red[0];
  __syncthreads();
  const float ex = valid ? expf(s - m) : 0.f;
  red[t] = ex;
  __syncthreads();
  for (int st = 64; st > 0; st >>= 1) {
    if (t < st) red[t] += red[t + st];
    __syncthreads();
  }
  sw[t] = ex / red[0];
  __syncthreads();
  float accv[8] = {0, 0, 0, 0, 0, 0, 0, 0};
  for (int tt = 0; tt < T_; ++tt) {
    const float wv = sw[tt];
    const float* hr = hs + (size_t)(b * T_ + tt) * EPROJS;
#pragma unroll
    for (int i = 0; i < 8; ++i) accv[i] += wv * hr[t + i * 128];
  }
#pragma unroll
  for (int i = 0; i < 8; ++i)
    att_c[(size_t)b * EPROJS + t + i * 128] = accv[i];
}

// x[b] = [ embed[ys[b,u]] (512) | att_c[b] (1024) ]
__global__ void build_x(const float* __restrict__ embed, const int* __restrict__ ys,
                        const float* __restrict__ att_c, float* __restrict__ x, int u) {
  const int b = blockIdx.x;
  const float* er = embed + (size_t)ys[b * U_ + u] * EMB;
  for (int j = threadIdx.x; j < EMB + EPROJS; j += blockDim.x)
    x[(size_t)b * (EMB + EPROJS) + j] = (j < EMB) ? er[j]
                                                  : att_c[(size_t)b * EPROJS + (j - EMB)];
}

// LSTM cell activations; optionally scatter h into h_dec[:,u,:]
__global__ void lstm_act(const float* __restrict__ g, float* __restrict__ h,
                         float* __restrict__ c, float* __restrict__ hdec_slot) {
  const int b = blockIdx.x;
  const float* gb = g + (size_t)b * (4 * DUNITS);
  for (int d = threadIdx.x; d < DUNITS; d += blockDim.x) {
    const float iv = sigf(gb[d]);
    const float fv = sigf(gb[DUNITS + d]);
    const float gv = tanhf(gb[2 * DUNITS + d]);
    const float ov = sigf(gb[3 * DUNITS + d]);
    const size_t si = (size_t)b * DUNITS + d;
    const float c2 = fv * c[si] + iv * gv;
    const float h2 = ov * tanhf(c2);
    c[si] = c2;
    h[si] = h2;
    if (hdec_slot) hdec_slot[(size_t)b * (U_ * DUNITS) + d] = h2;
  }
}

// ---------------------------------------------------------------------------
extern "C" void kernel_launch(void* const* d_in, const int* in_sizes, int n_in,
                              void* d_out, int out_size, void* d_ws, size_t ws_size,
                              hipStream_t stream) {
  (void)in_sizes; (void)n_in; (void)out_size; (void)ws_size;
  const float* hs_pad = (const float*)d_in[0];
  const int*   ys     = (const int*)d_in[1];
  const int*   hlens  = (const int*)d_in[2];
  const float* embed  = (const float*)d_in[3];
  const float* w_enc  = (const float*)d_in[4];
  const float* w_dec  = (const float*)d_in[5];
  const float* b_att  = (const float*)d_in[6];
  const float* v_att  = (const float*)d_in[7];
  const float* W_ih0  = (const float*)d_in[8];
  const float* W_hh0  = (const float*)d_in[9];
  const float* b0     = (const float*)d_in[10];
  const float* W_ih1  = (const float*)d_in[11];
  const float* W_hh1  = (const float*)d_in[12];
  const float* b1     = (const float*)d_in[13];
  const float* w_je   = (const float*)d_in[14];
  const float* b_je   = (const float*)d_in[15];
  const float* w_jd   = (const float*)d_in[16];
  const float* w_jo   = (const float*)d_in[17];
  const float* b_jo   = (const float*)d_in[18];
  float* out = (float*)d_out;

  char* ws = (char*)d_ws;
  size_t off = 0;
  auto take = [&](size_t bytes) -> void* {
    void* p = ws + off;
    off += (bytes + 255) & ~(size_t)255;
    return p;
  };
  // f16 weights ([N][K] layouts)
  _Float16* w_encT = (_Float16*)take((size_t)ATT_  * EPROJS * 2);
  _Float16* w_decT = (_Float16*)take((size_t)ATT_  * DUNITS * 2);
  _Float16* w_jeT  = (_Float16*)take((size_t)JOINT_ * EPROJS * 2);
  _Float16* w_jdT  = (_Float16*)take((size_t)JOINT_ * DUNITS * 2);
  _Float16* w_joT  = (_Float16*)take((size_t)ODIM  * JOINT_ * 2);
  _Float16* Wih0h  = (_Float16*)take((size_t)4 * DUNITS * (EMB + EPROJS) * 2);
  _Float16* Whh0h  = (_Float16*)take((size_t)4 * DUNITS * DUNITS * 2);
  _Float16* Wih1h  = (_Float16*)take((size_t)4 * DUNITS * DUNITS * 2);
  _Float16* Whh1h  = (_Float16*)take((size_t)4 * DUNITS * DUNITS * 2);
  // f32 intermediates
  float* pre_enc = (float*)take((size_t)B_ * T_ * ATT_ * 4);
  float* jeW     = (float*)take((size_t)B_ * T_ * JOINT_ * 4);
  float* jdW     = (float*)take((size_t)B_ * U_ * JOINT_ * 4);
  float* hdec    = (float*)take((size_t)B_ * U_ * DUNITS * 4);
  float* q       = (float*)take((size_t)B_ * ATT_ * 4);
  float* attc    = (float*)take((size_t)B_ * EPROJS * 4);
  float* x       = (float*)take((size_t)B_ * (EMB + EPROJS) * 4);
  float* g0      = (float*)take((size_t)B_ * 4 * DUNITS * 4);
  float* g1      = (float*)take((size_t)B_ * 4 * DUNITS * 4);
  float* states  = (float*)take((size_t)4 * B_ * DUNITS * 4);
  float* z0 = states;
  float* c0 = states + 1 * B_ * DUNITS;
  float* z1 = states + 2 * B_ * DUNITS;
  float* c1 = states + 3 * B_ * DUNITS;

  // ---- weight conversions -------------------------------------------------
  cvtT_f16<<<(EPROJS * ATT_) / 256, 256, 0, stream>>>(w_enc, w_encT, EPROJS, ATT_);
  cvtT_f16<<<(DUNITS * ATT_) / 256, 256, 0, stream>>>(w_dec, w_decT, DUNITS, ATT_);
  cvtT_f16<<<(EPROJS * JOINT_) / 256, 256, 0, stream>>>(w_je, w_jeT, EPROJS, JOINT_);
  cvtT_f16<<<(DUNITS * JOINT_) / 256, 256, 0, stream>>>(w_jd, w_jdT, DUNITS, JOINT_);
  cvtT_f16<<<(JOINT_ * ODIM) / 256, 256, 0, stream>>>(w_jo, w_joT, JOINT_, ODIM);
  cvt4_f16<<<(4 * DUNITS * (EMB + EPROJS)) / 1024, 256, 0, stream>>>(W_ih0, Wih0h);
  cvt4_f16<<<(4 * DUNITS * DUNITS) / 1024, 256, 0, stream>>>(W_hh0, Whh0h);
  cvt4_f16<<<(4 * DUNITS * DUNITS) / 1024, 256, 0, stream>>>(W_ih1, Wih1h);
  cvt4_f16<<<(4 * DUNITS * DUNITS) / 1024, 256, 0, stream>>>(W_hh1, Whh1h);

  // ---- encoder-side precomputes ------------------------------------------
  // pre_enc = hs_pad @ w_enc + b_att   : M=1024, N=512, K=1024
  gemm_a32_128x128<<<dim3(ATT_ / 128, (B_ * T_) / 128), 256, 0, stream>>>(
      hs_pad, w_encT, b_att, pre_enc, EPROJS, ATT_);
  // je = hs_pad @ w_je + b_je          : M=1024, N=1024, K=1024
  gemm_a32_128x128<<<dim3(JOINT_ / 128, (B_ * T_) / 128), 256, 0, stream>>>(
      hs_pad, w_jeT, b_je, jeW, EPROJS, JOINT_);

  // ---- decoder scan (sequential, 32 steps) -------------------------------
  hipMemsetAsync(states, 0, (size_t)4 * B_ * DUNITS * 4, stream);
  for (int u = 0; u < U_; ++u) {
    // q = z0 @ w_dec
    gemm_a32_16rows<<<ATT_ / 128, 256, 0, stream>>>(z0, w_decT, nullptr, q,
                                                    B_, DUNITS, ATT_, 0);
    att_kernel<<<B_, 128, 0, stream>>>(pre_enc, q, v_att, hs_pad, hlens, attc);
    build_x<<<B_, 256, 0, stream>>>(embed, ys, attc, x, u);
    // g0 = x @ W_ih0^T + b0 ; g0 += z0 @ W_hh0^T
    gemm_a32_16rows<<<(4 * DUNITS) / 128, 256, 0, stream>>>(x, Wih0h, b0, g0,
                                                            B_, EMB + EPROJS, 4 * DUNITS, 0);
    gemm_a32_16rows<<<(4 * DUNITS) / 128, 256, 0, stream>>>(z0, Whh0h, nullptr, g0,
                                                            B_, DUNITS, 4 * DUNITS, 1);
    lstm_act<<<B_, 256, 0, stream>>>(g0, z0, c0, nullptr);
    // g1 = h0 @ W_ih1^T + b1 ; g1 += z1 @ W_hh1^T
    gemm_a32_16rows<<<(4 * DUNITS) / 128, 256, 0, stream>>>(z0, Wih1h, b1, g1,
                                                            B_, DUNITS, 4 * DUNITS, 0);
    gemm_a32_16rows<<<(4 * DUNITS) / 128, 256, 0, stream>>>(z1, Whh1h, nullptr, g1,
                                                            B_, DUNITS, 4 * DUNITS, 1);
    lstm_act<<<B_, 256, 0, stream>>>(g1, z1, c1, hdec + (size_t)u * DUNITS);
  }

  // jd = h_dec @ w_jd : M=256, N=1024, K=1024
  gemm_a32_128x128<<<dim3(JOINT_ / 128, (B_ * U_) / 128), 256, 0, stream>>>(
      hdec, w_jdT, nullptr, jdW, DUNITS, JOINT_);

  // out = tanh(je + jd) @ w_jo + b_jo : M=32768, N=4096, K=1024
  joint_out_gemm<<<dim3(ODIM / 128, (B_ * T_ * U_) / 128), 256, 0, stream>>>(
      jeW, jdW, w_joT, b_jo, out);
}
